// test_target_18700287606953
// MI455X (gfx1250) — matrix operand order repaired and checked by reading, not compiled, not run on hardware
//
#include <hip/hip_runtime.h>
#include <hip/hip_bf16.h>
#include <math.h>

// ---- problem constants (match reference) ----
#define BATCH 128
#define SEQ   40
#define EMB   512
#define HID   1024
#define FOURH 4096
#define VOC   32000
#define IN1   21504   // SEQ*EMB + 2*EMB

typedef __bf16 v16bf __attribute__((ext_vector_type(16)));
typedef float  v8f   __attribute__((ext_vector_type(8)));

union BFV { v16bf v; unsigned short s[16]; };

// fp32 -> bf16 bits, round-to-nearest-even
__device__ __forceinline__ unsigned short f2bf(float f) {
  union { float f; unsigned int u; } x; x.f = f;
  unsigned int r = x.u + 0x7FFFu + ((x.u >> 16) & 1u);
  return (unsigned short)(r >> 16);
}

// LDS B-tile layout: halfword index = t*640 + col*40 + k'
//   t   = K-subtile (0,1) covering k' 0..31
//   col = 0..15 (N within tile), stride 40 halfwords (80 B) -> bank-conflict-free
//         fragment reads and 16-B aligned for uint4 LDS loads.
#define BS_WORDS 1280

// Accumulate 16(M)x16(N) per wave of A(bf16 row-major MxK) * W(fp32 row-major KxN).
// Block = 256 threads: waves cover M=128, block covers one 16-col N tile.
// Weights staged once per block through LDS as bf16 in WMMA fragment order.
// Activations are pre-converted bf16 -> A frag = two b128 loads, no VALU.
__device__ __forceinline__ void gemm_phase(const unsigned short* __restrict__ A, int K,
                                           const float* __restrict__ W, int N,
                                           int mbase, int nbase, int lhalf, int l15,
                                           int tid, unsigned short* bs, v8f& acc) {
  const int row = tid >> 2;        // 0..63 : K row within 64-chunk
  const int cg  = (tid & 3) << 2;  // 0,4,8,12 : starting col of float4
  const unsigned short* arow = A + (long)(mbase + l15) * K;

  for (int k0 = 0; k0 < K; k0 += 64) {
    __syncthreads();  // previous chunk's fragment reads complete
    {
      // cooperative load 64(K) x 16(N) fp32 weight tile, coalesced 64B per K-row
      const float* src = W + (long)(k0 + row) * N + (nbase + cg);
      float4 w4 = *(const float4*)src;
      if (k0 + 64 < K)  // prefetch next K-chunk (global_prefetch_b8)
        __builtin_prefetch(src + (long)64 * N, 0, 1);
      const int tt = row >> 5, kp = row & 31;
      unsigned short* dst = bs + tt * 640 + kp;
      dst[(cg + 0) * 40] = f2bf(w4.x);
      dst[(cg + 1) * 40] = f2bf(w4.y);
      dst[(cg + 2) * 40] = f2bf(w4.z);
      dst[(cg + 3) * 40] = f2bf(w4.w);
    }
    __syncthreads();  // staging visible to all waves

#pragma unroll
    for (int t = 0; t < 2; ++t) {
      BFV a, b;
      // A 16x32 bf16 frag: lanes<16 hold K {0..7,16..23}, lanes>=16 {8..15,24..31}
      const int ka = k0 + t * 32 + lhalf * 8;
      *(uint4*)&a.s[0] = *(const uint4*)(arow + ka);        // K ka..ka+7
      *(uint4*)&a.s[8] = *(const uint4*)(arow + ka + 16);   // K ka+16..ka+23
      // B 32x16 bf16 frag: lane = column, K {0..7,16..23} / {8..15,24..31} from LDS
      const unsigned short* colp = bs + t * 640 + l15 * 40 + lhalf * 8;
      *(uint4*)&b.s[0] = *(const uint4*)colp;
      *(uint4*)&b.s[8] = *(const uint4*)(colp + 16);
      acc = __builtin_amdgcn_wmma_f32_16x16x32_bf16(false, a.v, false, b.v,
                                                    (short)0, acc, false, false);
    }
  }
}

// Z[128 x N] = A1[128 x K1] @ W1 (+ A2[128 x K2] @ W2) + bias   (A* are bf16)
__global__ __launch_bounds__(256) void gemm_wmma(
    const unsigned short* __restrict__ A1, int K1, const float* __restrict__ W1,
    const unsigned short* __restrict__ A2, int K2, const float* __restrict__ W2,
    const float* __restrict__ bias, float* __restrict__ Z, int N) {
  __shared__ __align__(16) unsigned short bs[BS_WORDS];
  const int tid   = threadIdx.x;
  const int lane  = tid & 31;
  const int wave  = tid >> 5;
  const int mbase = wave * 16;
  const int nbase = blockIdx.x * 16;
  const int lhalf = lane >> 4;   // 0: lanes 0-15, 1: lanes 16-31
  const int l15   = lane & 15;

  v8f acc = {};
  gemm_phase(A1, K1, W1, N, mbase, nbase, lhalf, l15, tid, bs, acc);
  if (A2) gemm_phase(A2, K2, W2, N, mbase, nbase, lhalf, l15, tid, bs, acc);

  const float bv = bias ? bias[nbase + l15] : 0.0f;
  // C/D layout: VGPR r -> M = r (lanes 0-15) or r+8 (lanes 16-31), N = lane&15
#pragma unroll
  for (int r = 0; r < 8; ++r) {
    int m = mbase + r + lhalf * 8;
    Z[(long)m * N + nbase + l15] = acc[r] + bv;
  }
}

// x_bf16 = [enc_flat | word_embedding | persona_embedding[speaker]]  (128 x 21504)
__global__ void build_x(const float* __restrict__ enc, const float* __restrict__ we,
                        const int* __restrict__ spk, const float* __restrict__ pe,
                        unsigned short* __restrict__ x) {
  const int total = BATCH * IN1;
  for (int idx = blockIdx.x * blockDim.x + threadIdx.x; idx < total;
       idx += gridDim.x * blockDim.x) {
    int b = idx / IN1, k = idx - b * IN1;
    float v;
    if (k < SEQ * EMB)            v = enc[b * (SEQ * EMB) + k];
    else if (k < SEQ * EMB + EMB) v = we[b * EMB + (k - SEQ * EMB)];
    else                          v = pe[spk[b] * EMB + (k - SEQ * EMB - EMB)];
    x[idx] = f2bf(v);
  }
}

// fp32 -> bf16 bulk convert (for h0)
__global__ void cvt_bf16(const float* __restrict__ src, unsigned short* __restrict__ dst,
                         int n) {
  int i = blockIdx.x * blockDim.x + threadIdx.x;
  if (i < n) dst[i] = f2bf(src[i]);
}

// Keras LSTM gates, order i,f,g,o; activation = relu.
// h is emitted in bf16 (feeds next GEMM); final layer also emits fp32 h4/c4.
__global__ __launch_bounds__(256) void lstm_gates(
    const float* __restrict__ Z, const float* __restrict__ c_prev,
    unsigned short* __restrict__ h_out_bf, float* __restrict__ c_out,
    float* __restrict__ h_out_f32, float* __restrict__ c_out_f32) {
  int idx = blockIdx.x * blockDim.x + threadIdx.x;
  if (idx >= BATCH * HID) return;
  int b = idx / HID, j = idx - b * HID;
  const float* z = Z + (long)b * FOURH;
  float zi = z[j], zf = z[HID + j], zg = z[2 * HID + j], zo = z[3 * HID + j];
  float si = 1.0f / (1.0f + expf(-zi));
  float sf = 1.0f / (1.0f + expf(-zf));
  float so = 1.0f / (1.0f + expf(-zo));
  float g  = fmaxf(zg, 0.0f);
  float cn = sf * c_prev[idx] + si * g;
  float hn = so * fmaxf(cn, 0.0f);
  h_out_bf[idx] = f2bf(hn);
  c_out[idx] = cn;
  if (h_out_f32) { h_out_f32[idx] = hn; c_out_f32[idx] = cn; }
}

// in-place row softmax over VOC columns; one block per batch row
__global__ __launch_bounds__(256) void softmax_inplace(float* __restrict__ logits) {
  __shared__ float red[256];
  const int t = threadIdx.x;
  float* row = logits + (long)blockIdx.x * VOC;

  float m = -3.4e38f;
  for (int j = t; j < VOC; j += 256) m = fmaxf(m, row[j]);
  red[t] = m; __syncthreads();
  for (int s = 128; s > 0; s >>= 1) {
    if (t < s) red[t] = fmaxf(red[t], red[t + s]);
    __syncthreads();
  }
  m = red[0]; __syncthreads();

  float sum = 0.0f;
  for (int j = t; j < VOC; j += 256) {
    float e = expf(row[j] - m);
    row[j] = e;
    sum += e;
  }
  red[t] = sum; __syncthreads();
  for (int s = 128; s > 0; s >>= 1) {
    if (t < s) red[t] += red[t + s];
    __syncthreads();
  }
  float inv = 1.0f / red[0];

  for (int j = t; j < VOC; j += 256) row[j] *= inv;
}

extern "C" void kernel_launch(void* const* d_in, const int* in_sizes, int n_in,
                              void* d_out, int out_size, void* d_ws, size_t ws_size,
                              hipStream_t stream) {
  (void)in_sizes; (void)n_in; (void)out_size; (void)ws_size;
  const float* enc = (const float*)d_in[0];
  const float* h0  = (const float*)d_in[1];
  const float* c0  = (const float*)d_in[2];
  const float* we  = (const float*)d_in[3];
  const int*   spk = (const int*)d_in[4];
  const float* pe  = (const float*)d_in[5];
  const float* W[4]  = {(const float*)d_in[6],  (const float*)d_in[9],
                        (const float*)d_in[12], (const float*)d_in[15]};
  const float* U[4]  = {(const float*)d_in[7],  (const float*)d_in[10],
                        (const float*)d_in[13], (const float*)d_in[16]};
  const float* bb[4] = {(const float*)d_in[8],  (const float*)d_in[11],
                        (const float*)d_in[14], (const float*)d_in[17]};
  const float* Wd = (const float*)d_in[18];
  const float* bd = (const float*)d_in[19];
  // d_in[20] = is_first_word: reshape is a no-op on flat memory, ignore.

  float* out    = (float*)d_out;
  float* probs  = out;                          // 128*32000
  float* h4_out = out + (long)BATCH * VOC;      // 128*1024
  float* c4_out = h4_out + BATCH * HID;         // 128*1024

  // workspace layout (16B alignment maintained at every boundary)
  unsigned short* xb  = (unsigned short*)d_ws;                 // 128*21504 bf16
  float*          z   = (float*)(xb + (long)BATCH * IN1);      // 128*4096  f32
  unsigned short* hb  = (unsigned short*)(z + (long)BATCH * FOURH); // 128*1024 bf16
  unsigned short* h0b = hb + (long)BATCH * HID;                // 128*1024 bf16
  float*          c   = (float*)(h0b + (long)BATCH * HID);     // 128*1024 f32

  build_x<<<256, 256, 0, stream>>>(enc, we, spk, pe, xb);
  cvt_bf16<<<(BATCH * HID) / 256, 256, 0, stream>>>(h0, h0b, BATCH * HID);

  // layer 1: z = x@W1 + h0@U1 + b1
  gemm_wmma<<<FOURH / 16, 256, 0, stream>>>(xb, IN1, W[0], h0b, HID, U[0], bb[0], z, FOURH);
  lstm_gates<<<(BATCH * HID) / 256, 256, 0, stream>>>(z, c0, hb, c, nullptr, nullptr);

  // layers 2..4: z = h@Wl + h@Ul + bl   (input == hidden state == prev layer output)
  for (int l = 1; l < 4; ++l) {
    gemm_wmma<<<FOURH / 16, 256, 0, stream>>>(hb, HID, W[l], hb, HID, U[l], bb[l], z, FOURH);
    const bool last = (l == 3);
    lstm_gates<<<(BATCH * HID) / 256, 256, 0, stream>>>(
        z, c, hb, c, last ? h4_out : nullptr, last ? c4_out : nullptr);
  }

  // decoder: logits = h4 @ Wd + bd  (written straight into d_out), then softmax in place
  gemm_wmma<<<VOC / 16, 256, 0, stream>>>(hb, HID, Wd, nullptr, 0, nullptr, bd, probs, VOC);
  softmax_inplace<<<BATCH, 256, 0, stream>>>(probs);
}
